// ProbSparseAttention_17841294148203
// MI455X (gfx1250) — compile-verified, matrix operand order checked by reading, not run on hardware
//
#include <hip/hip_runtime.h>
#include <math.h>

#define B_ 2
#define L_ 2048
#define H_ 8
#define D_ 64
#define U_ 80

typedef __attribute__((ext_vector_type(16))) __bf16 v16bf;
typedef __attribute__((ext_vector_type(8)))  float  v8f;
typedef __attribute__((ext_vector_type(4)))  unsigned int v4u;
typedef __attribute__((ext_vector_type(8)))  int v8i_;
typedef __attribute__((ext_vector_type(4)))  int v4i_;

union BfFrag {
    v16bf v;
    unsigned short u[16];
};

// float -> bf16 (round to nearest even; inputs finite)
__device__ __forceinline__ unsigned short f2bf(float f) {
    unsigned int x = __float_as_uint(f);
    x += 0x7FFFu + ((x >> 16) & 1u);
    return (unsigned short)(x >> 16);
}

// K-index pattern for 16-bit A/B fragments (ISA 7.12.2, wave32):
// lanes 0-15: h0..7 -> K=0..7,  h8..15 -> K=16..23
// lanes 16-31: h0..7 -> K=8..15, h8..15 -> K=24..31
__device__ __forceinline__ int akd(int lane, int h) {
    int klo = (lane & 16) ? 8 : 0;
    return (h < 8) ? (klo + h) : (16 + klo + (h - 8));
}

// ---------------------------------------------------------------------------
// TDM 2D tile load: 32 rows x 64 bf16 elements, contiguous tensor (stride 64).
// D# per ISA ch.8: group0 = {count, lds_addr, global_addr, type=2},
// group1 = {data_size=2B, tensor_dim0=64, tensor_dim1=2048, tile 64x32, stride 64}.
// Groups 2/3 zero (2D tile, tile_dim2..4 = 0). 6-arg toolchain form.
// ---------------------------------------------------------------------------
__device__ __forceinline__ void tdm_load_tile_2d(unsigned lds_byte_off,
                                                 const unsigned short* gptr) {
    unsigned long long ga = (unsigned long long)gptr;
    v4u g0;
    g0[0] = 1u;                                                  // count=1, user mode
    g0[1] = lds_byte_off;                                        // lds_addr (bytes)
    g0[2] = (unsigned)ga;                                        // global_addr[31:0]
    g0[3] = ((unsigned)(ga >> 32) & 0x01FFFFFFu) | 0x80000000u;  // [56:32], type=2
    v8i_ g1;
    g1[0] = 1 << 16;                  // wg_mask=0, data_size=1 (2 bytes)
    g1[1] = (int)(64u << 16);         // tensor_dim0[15:0] = 64 (bits 63:48)
    g1[2] = (int)(2048u << 16);       // tensor_dim0 hi=0, tensor_dim1[15:0]=2048
    g1[3] = (int)(64u << 16);         // tensor_dim1 hi=0, tile_dim0=64
    g1[4] = 32;                       // tile_dim1=32, tile_dim2=0
    g1[5] = 64;                       // tensor_dim0_stride = 64 (elements)
    g1[6] = 0;                        // stride hi / tensor_dim1_stride = 0
    g1[7] = 0;
    v4i_ z4 = {0, 0, 0, 0};
    v8i_ z8 = {0, 0, 0, 0, 0, 0, 0, 0};
    __builtin_amdgcn_tensor_load_to_lds(g0, g1, z4, z4, z8, 0);
}

// ---------------------------------------------------------------------------
// Kernel 0: convert K,V f32 (B,L,H,D) -> bf16 workspace, layout [b,h][l][d]
// so each 32-key chunk is one contiguous 4KB TDM tile.
// ---------------------------------------------------------------------------
__global__ void __launch_bounds__(256)
convert_kv_kernel(const float* __restrict__ K, const float* __restrict__ V,
                  unsigned short* __restrict__ Kbf, unsigned short* __restrict__ Vbf) {
    int g = blockIdx.x * blockDim.x + threadIdx.x;   // over B*H*L*D = 2^21
    if (g >= B_ * H_ * L_ * D_) return;
    int d = g & (D_ - 1);
    int l = (g >> 6) & (L_ - 1);
    int h = (g >> 17) & (H_ - 1);
    int b = g >> 20;
    size_t src = (((size_t)b * L_ + l) * H_ + h) * D_ + d;
    Kbf[g] = f2bf(K[src]);
    Vbf[g] = f2bf(V[src]);
}

// ---------------------------------------------------------------------------
// Kernel 1: sparsity measure M[b,h,l] = max_s(q.k_s) - sum_s(q.k_s)/L
// ---------------------------------------------------------------------------
__global__ void __launch_bounds__(256)
compute_m_kernel(const float* __restrict__ Q, const float* __restrict__ K,
                 const int* __restrict__ samp, float* __restrict__ Mout) {
    int g = blockIdx.x * blockDim.x + threadIdx.x;
    if (g >= B_ * H_ * L_) return;
    int l = g & (L_ - 1);
    int h = (g >> 11) & (H_ - 1);
    int b = g >> 14;
    const float* q = Q + (((size_t)b * L_ + l) * H_ + h) * D_;
    float qv[D_];
#pragma unroll
    for (int d = 0; d < D_; ++d) qv[d] = q[d];
    float mx = -1e30f, sm = 0.f;
    for (int s = 0; s < U_; ++s) {
        int kk = samp[l * U_ + s];
        const float* kr = K + (((size_t)b * L_ + kk) * H_ + h) * D_;
        float acc = 0.f;
#pragma unroll
        for (int d = 0; d < D_; ++d) acc += qv[d] * kr[d];
        mx = fmaxf(mx, acc);
        sm += acc;
    }
    Mout[((size_t)b * H_ + h) * L_ + l] = mx - sm * (1.0f / (float)L_);
}

// ---------------------------------------------------------------------------
// Kernel 2: top-80 selection per (b,h), ties -> lower index (like lax.top_k)
// ---------------------------------------------------------------------------
__global__ void __launch_bounds__(256)
topk_kernel(const float* __restrict__ M, int* __restrict__ mtop) {
    __shared__ float vals[L_];
    __shared__ float rv[256];
    __shared__ int   ri[256];
    int bh = blockIdx.x;
    int tid = threadIdx.x;
    for (int i = tid; i < L_; i += 256) vals[i] = M[(size_t)bh * L_ + i];
    __syncthreads();
    for (int u = 0; u < U_; ++u) {
        float best = -1e30f;
        int   bi = L_;
        for (int i = tid; i < L_; i += 256) {
            float v = vals[i];
            if (v > best || (v == best && i < bi)) { best = v; bi = i; }
        }
        rv[tid] = best; ri[tid] = bi;
        __syncthreads();
        for (int s = 128; s > 0; s >>= 1) {
            if (tid < s) {
                if (rv[tid + s] > rv[tid] ||
                    (rv[tid + s] == rv[tid] && ri[tid + s] < ri[tid])) {
                    rv[tid] = rv[tid + s]; ri[tid] = ri[tid + s];
                }
            }
            __syncthreads();
        }
        if (tid == 0) { mtop[bh * U_ + u] = ri[0]; vals[ri[0]] = -1e30f; }
        __syncthreads();
    }
}

// ---------------------------------------------------------------------------
// Kernel 3: flash-attention over selected queries (one wave/block, WMMA bf16).
// K/V chunks (32 keys x 64 d, bf16) streamed global->LDS by the Tensor Data
// Mover, double-buffered, synchronized with s_wait_tensorcnt.
// ---------------------------------------------------------------------------
__global__ void __launch_bounds__(32)
attn_kernel(const float* __restrict__ Q,
            const unsigned short* __restrict__ Kbf,
            const unsigned short* __restrict__ Vbf,
            const int* __restrict__ mtop, float* __restrict__ upd) {
    __shared__ unsigned short klds[2][32 * 64];  // [buf][key*64 + d]
    __shared__ unsigned short vlds[2][32 * 64];
    __shared__ unsigned short plds[16 * 32];     // [row*32 + key]

    int blk  = blockIdx.x;          // 0..79
    int tile = blk % (U_ / 16);     // 0..4
    int bh   = blk / (U_ / 16);     // 0..15
    int h    = bh & (H_ - 1);
    int b    = bh >> 3;
    int lane = threadIdx.x;         // 0..31
    int col  = lane & 15;
    int hi8  = (lane & 16) ? 8 : 0;

    // LDS byte offsets (generic LDS pointers carry the offset in addr[31:0])
    unsigned k_off[2], v_off[2];
    k_off[0] = (unsigned)(unsigned long long)(void*)&klds[0][0];
    k_off[1] = (unsigned)(unsigned long long)(void*)&klds[1][0];
    v_off[0] = (unsigned)(unsigned long long)(void*)&vlds[0][0];
    v_off[1] = (unsigned)(unsigned long long)(void*)&vlds[1][0];

    const unsigned short* kg = Kbf + (size_t)bh * L_ * D_;
    const unsigned short* vg = Vbf + (size_t)bh * L_ * D_;

    // Q A-fragments, pre-scaled by 1/sqrt(D) (softmax is shift-invariant,
    // scale folds into the QK^T product)
    int qi = mtop[bh * U_ + tile * 16 + col];
    const float* qptr = Q + (((size_t)b * L_ + qi) * H_ + h) * D_;
    BfFrag aq0, aq1;
#pragma unroll
    for (int hh = 0; hh < 16; ++hh) {
        int d = akd(lane, hh);
        aq0.u[hh] = f2bf(0.125f * qptr[d]);
        aq1.u[hh] = f2bf(0.125f * qptr[32 + d]);
    }

    v8f o0 = {}, o1 = {}, o2 = {}, o3 = {};
    float rowm[8], rowl[8];
#pragma unroll
    for (int i = 0; i < 8; ++i) { rowm[i] = -1e30f; rowl[i] = 0.f; }

    // prologue: chunk 0 into buffer 0 (2 TDM ops in flight)
    tdm_load_tile_2d(k_off[0], kg);
    tdm_load_tile_2d(v_off[0], vg);

    for (int kb = 0; kb < L_ / 32; ++kb) {
        int cur = kb & 1;
        if (kb + 1 < L_ / 32) {
            int nxt = cur ^ 1;
            const unsigned short* kn = kg + (size_t)(kb + 1) * 32 * D_;
            const unsigned short* vn = vg + (size_t)(kb + 1) * 32 * D_;
            tdm_load_tile_2d(k_off[nxt], kn);
            tdm_load_tile_2d(v_off[nxt], vn);
            __builtin_amdgcn_s_wait_tensorcnt((short)2);  // chunk kb landed
        } else {
            __builtin_amdgcn_s_wait_tensorcnt((short)0);
        }
        const unsigned short* kl = klds[cur];
        const unsigned short* vl = vlds[cur];

        // S(16x32) = (Q*scale)(16x64) . K_chunk^T : two 16x16 tiles, K-dim 64
        v8f s0 = {}, s1 = {};
        {
            BfFrag b0lo, b0hi, b1lo, b1hi;
#pragma unroll
            for (int hh = 0; hh < 16; ++hh) {
                int d = akd(lane, hh);
                b0lo.u[hh] = kl[col * 64 + d];
                b0hi.u[hh] = kl[col * 64 + 32 + d];
                b1lo.u[hh] = kl[(16 + col) * 64 + d];
                b1hi.u[hh] = kl[(16 + col) * 64 + 32 + d];
            }
            s0 = __builtin_amdgcn_wmma_f32_16x16x32_bf16(false, aq0.v, false, b0lo.v, (short)0, s0, false, false);
            s0 = __builtin_amdgcn_wmma_f32_16x16x32_bf16(false, aq1.v, false, b0hi.v, (short)0, s0, false, false);
            s1 = __builtin_amdgcn_wmma_f32_16x16x32_bf16(false, aq0.v, false, b1lo.v, (short)0, s1, false, false);
            s1 = __builtin_amdgcn_wmma_f32_16x16x32_bf16(false, aq1.v, false, b1hi.v, (short)0, s1, false, false);
        }

        // online softmax (row stats replicated across the 16 lanes of each half)
#pragma unroll
        for (int i = 0; i < 8; ++i) {
            float v0 = s0[i], v1 = s1[i];
            float mloc = fmaxf(v0, v1);
#pragma unroll
            for (int m = 1; m < 16; m <<= 1) mloc = fmaxf(mloc, __shfl_xor(mloc, m, 32));
            float mnew = fmaxf(rowm[i], mloc);
            float corr = __expf(rowm[i] - mnew);
            rowm[i] = mnew;
            float p0 = __expf(v0 - mnew);
            float p1 = __expf(v1 - mnew);
            float rs = p0 + p1;
#pragma unroll
            for (int m = 1; m < 16; m <<= 1) rs += __shfl_xor(rs, m, 32);
            rowl[i] = rowl[i] * corr + rs;
            o0[i] *= corr; o1[i] *= corr; o2[i] *= corr; o3[i] *= corr;
            int row = i + hi8;
            plds[row * 32 + col]      = f2bf(p0);
            plds[row * 32 + 16 + col] = f2bf(p1);
        }
        __syncthreads();

        // O(16x64) += P(16x32) . V_chunk(32x64) : 4 output tiles, K=32
        BfFrag pa;
#pragma unroll
        for (int hh = 0; hh < 16; ++hh) pa.u[hh] = plds[col * 32 + akd(lane, hh)];
        {
            BfFrag vb0, vb1, vb2, vb3;
#pragma unroll
            for (int hh = 0; hh < 16; ++hh) {
                int k = akd(lane, hh);
                vb0.u[hh] = vl[k * 64 + col];
                vb1.u[hh] = vl[k * 64 + 16 + col];
                vb2.u[hh] = vl[k * 64 + 32 + col];
                vb3.u[hh] = vl[k * 64 + 48 + col];
            }
            o0 = __builtin_amdgcn_wmma_f32_16x16x32_bf16(false, pa.v, false, vb0.v, (short)0, o0, false, false);
            o1 = __builtin_amdgcn_wmma_f32_16x16x32_bf16(false, pa.v, false, vb1.v, (short)0, o1, false, false);
            o2 = __builtin_amdgcn_wmma_f32_16x16x32_bf16(false, pa.v, false, vb2.v, (short)0, o2, false, false);
            o3 = __builtin_amdgcn_wmma_f32_16x16x32_bf16(false, pa.v, false, vb3.v, (short)0, o3, false, false);
        }
        __syncthreads();
    }

    // normalize and write upd[(bh, j, d)] with j = tile*16 + row
#pragma unroll
    for (int i = 0; i < 8; ++i) {
        float inv = 1.0f / rowl[i];
        int row = i + hi8;
        size_t base = ((size_t)bh * U_ + tile * 16 + row) * D_;
        upd[base + col]      = o0[i] * inv;
        upd[base + 16 + col] = o1[i] * inv;
        upd[base + 32 + col] = o2[i] * inv;
        upd[base + 48 + col] = o3[i] * inv;
    }
}

// ---------------------------------------------------------------------------
// Kernel 4: V mean over L per (b,h,d)
// ---------------------------------------------------------------------------
__global__ void __launch_bounds__(64)
vmean_kernel(const float* __restrict__ V, float* __restrict__ vmean) {
    int bh = blockIdx.x;
    int d  = threadIdx.x;
    int h = bh & (H_ - 1), b = bh >> 3;
    float s = 0.f;
    for (int l = 0; l < L_; ++l)
        s += V[(((size_t)b * L_ + l) * H_ + h) * D_ + d];
    vmean[(size_t)bh * D_ + d] = s * (1.0f / (float)L_);
}

// ---------------------------------------------------------------------------
// Kernel 5: context = broadcast(V_mean); scatter upd at clip(M_top,0,79),
// ascending j => last-wins, output layout (B, u, H, D)
// ---------------------------------------------------------------------------
__global__ void __launch_bounds__(64)
finalize_kernel(const float* __restrict__ vmean, const float* __restrict__ upd,
                const int* __restrict__ mtop, float* __restrict__ out) {
    int bh = blockIdx.x;
    int d  = threadIdx.x;
    int h = bh & (H_ - 1), b = bh >> 3;
    float vm = vmean[(size_t)bh * D_ + d];
    for (int u = 0; u < U_; ++u)
        out[(((size_t)b * U_ + u) * H_ + h) * D_ + d] = vm;
    for (int j = 0; j < U_; ++j) {
        int r = mtop[bh * U_ + j];
        r = r < 0 ? 0 : (r > U_ - 1 ? U_ - 1 : r);
        out[(((size_t)b * U_ + r) * H_ + h) * D_ + d] =
            upd[((size_t)bh * U_ + j) * D_ + d];
    }
}

// ---------------------------------------------------------------------------
extern "C" void kernel_launch(void* const* d_in, const int* in_sizes, int n_in,
                              void* d_out, int out_size, void* d_ws, size_t ws_size,
                              hipStream_t stream) {
    (void)in_sizes; (void)n_in; (void)out_size; (void)ws_size;
    const float* Q    = (const float*)d_in[0];
    const float* K    = (const float*)d_in[1];
    const float* V    = (const float*)d_in[2];
    const int*   samp = (const int*)d_in[3];
    float* out = (float*)d_out;

    // workspace: M | mtop | vmean | upd | Kbf | Vbf   (~8.9 MB)
    float* Mws   = (float*)d_ws;                            // 32768 f32
    int*   mtop  = (int*)(Mws + B_ * H_ * L_);              // 1280  i32
    float* vmean = (float*)(mtop + B_ * H_ * U_);           // 1024  f32
    float* upd   = vmean + B_ * H_ * D_;                    // 81920 f32
    unsigned short* Kbf = (unsigned short*)(upd + (size_t)B_ * H_ * U_ * D_); // 2M bf16
    unsigned short* Vbf = Kbf + (size_t)B_ * H_ * L_ * D_;                    // 2M bf16

    convert_kv_kernel<<<(B_ * H_ * L_ * D_ + 255) / 256, 256, 0, stream>>>(K, V, Kbf, Vbf);
    compute_m_kernel<<<(B_ * H_ * L_ + 255) / 256, 256, 0, stream>>>(Q, K, samp, Mws);
    topk_kernel<<<B_ * H_, 256, 0, stream>>>(Mws, mtop);
    attn_kernel<<<B_ * H_ * (U_ / 16), 32, 0, stream>>>(Q, Kbf, Vbf, mtop, upd);
    vmean_kernel<<<B_ * H_, D_, 0, stream>>>(V, vmean);
    finalize_kernel<<<B_ * H_, D_, 0, stream>>>(vmean, upd, mtop, out);
}
